// CodebookAttention_50525995270580
// MI455X (gfx1250) — matrix operand order repaired and checked by reading, not compiled, not run on hardware
//
#include <hip/hip_runtime.h>
#include <math.h>

#define B_   4
#define S_   4096
#define H_   2048
#define NH_  8
#define D_   256
#define CB_  2048
#define SEL_ 64
#define Q_   512

typedef __attribute__((ext_vector_type(16))) __bf16 v16bf;
typedef __attribute__((ext_vector_type(8)))  float  v8f;

union FragU { unsigned int u[8]; v16bf v; };

__device__ __forceinline__ unsigned short f2bf(float f) {
  unsigned int u = __float_as_uint(f);
  u += 0x7fffu + ((u >> 16) & 1u);           // RNE
  return (unsigned short)(u >> 16);
}
__device__ __forceinline__ float bf2f(unsigned short u) {
  return __uint_as_float(((unsigned int)u) << 16);
}
__device__ __forceinline__ float gelu_exact(float x) {
  return 0.5f * x * (1.0f + erff(x * 0.70710678118654752f));
}

__device__ __forceinline__ float block_reduce_sum(float v, float* red) {
  int t = threadIdx.x;
  red[t] = v; __syncthreads();
  for (int s = 128; s > 0; s >>= 1) { if (t < s) red[t] += red[t + s]; __syncthreads(); }
  float r = red[0]; __syncthreads();
  return r;
}
__device__ __forceinline__ float block_reduce_max(float v, float* red) {
  int t = threadIdx.x;
  red[t] = v; __syncthreads();
  for (int s = 128; s > 0; s >>= 1) { if (t < s) red[t] = fmaxf(red[t], red[t + s]); __syncthreads(); }
  float r = red[0]; __syncthreads();
  return r;
}

// CDNA5 async global->LDS copy (ASYNCcnt) --------------------------------
__device__ __forceinline__ void async_b128(unsigned lds_off, unsigned long long gaddr) {
  asm volatile("global_load_async_to_lds_b128 %0, %1, off"
               :: "v"(lds_off), "v"(gaddr) : "memory");
}
__device__ __forceinline__ void wait_async0() {
  asm volatile("s_wait_asynccnt 0" ::: "memory");
}

// ---------------------------------------------------------------------------
// WMMA GEMM:  C[M,N] = act( A[M,K](bf16) * W[N,K](bf16)^T + bias[N] )
// Block: 256 threads / 8 waves; tile 64x128; BK=32; 2x2 WMMA per wave;
// double-buffered LDS fed by global_load_async_to_lds_b128.
// M multiple of 64, N multiple of 128, K multiple of 32.
// ---------------------------------------------------------------------------
#define BM   64
#define BNt  128
#define BK   32
#define LDSP 40   // padded ushort row stride: 80B rows (16B-aligned, low-conflict)

__global__ __launch_bounds__(256) void wmma_gemm_bf16(
    const unsigned short* __restrict__ A, const unsigned short* __restrict__ W,
    const float* __restrict__ bias,
    float* __restrict__ Cf, unsigned short* __restrict__ Cb,
    int M, int N, int K, int act)
{
  __shared__ unsigned short sA[2][BM * LDSP];
  __shared__ unsigned short sB[2][BNt * LDSP];

  const int tid  = threadIdx.x;
  const int lane = tid & 31;
  const int wave = tid >> 5;
  const int wn   = wave & 3;    // n group: 32 cols
  const int wm   = wave >> 2;   // m group: 32 rows
  const int m0   = blockIdx.y * BM;
  const int n0   = blockIdx.x * BNt;
  const int half = lane >> 4;
  const int ln16 = lane & 15;

  // loader mapping: A tile 64x32 (16B/thread), B tile 128x32 (32B/thread)
  const int arow = tid >> 2, acol = (tid & 3) * 8;
  const int brow = tid >> 1, bcol = (tid & 1) * 16;

  v8f acc[2][2] = {};

  auto issue = [&](int k0, int buf) {
    const unsigned short* ga = A + (size_t)(m0 + arow) * K + k0 + acol;
    unsigned la = (unsigned)(size_t)&sA[buf][arow * LDSP + acol];
    async_b128(la, (unsigned long long)(size_t)ga);
    const unsigned short* gb = W + (size_t)(n0 + brow) * K + k0 + bcol;
    unsigned lb = (unsigned)(size_t)&sB[buf][brow * LDSP + bcol];
    async_b128(lb,      (unsigned long long)(size_t)gb);
    async_b128(lb + 16, (unsigned long long)(size_t)(gb + 8));
  };

  const int nk = K / BK;
  issue(0, 0);
  wait_async0();
  __syncthreads();

  for (int i = 0; i < nk; ++i) {
    const int cur = i & 1;
    if (i + 1 < nk) issue((i + 1) * BK, cur ^ 1);

    const unsigned short* pA = sA[cur];
    const unsigned short* pB = sB[cur];

    FragU fa[2], fb[2];
#pragma unroll
    for (int mt = 0; mt < 2; ++mt)
#pragma unroll
      for (int v = 0; v < 8; ++v) {
        int kk = (v < 4 ? 2 * v : 16 + 2 * (v - 4)) + half * 8;   // A 16x32 layout
        fa[mt].u[v] = *(const unsigned int*)&pA[(wm * 32 + mt * 16 + ln16) * LDSP + kk];
      }
#pragma unroll
    for (int nt = 0; nt < 2; ++nt)
#pragma unroll
      for (int v = 0; v < 8; ++v) {
        int kk = (v < 4 ? 2 * v : 16 + 2 * (v - 4)) + half * 8;   // B 32x16 layout
        fb[nt].u[v] = *(const unsigned int*)&pB[(wn * 32 + nt * 16 + ln16) * LDSP + kk];
      }
#pragma unroll
    for (int mt = 0; mt < 2; ++mt)
#pragma unroll
      for (int nt = 0; nt < 2; ++nt)
        acc[mt][nt] = __builtin_amdgcn_wmma_f32_16x16x32_bf16(
            false, fa[mt].v, false, fb[nt].v, (short)0, acc[mt][nt], false, false);

    wait_async0();
    __syncthreads();
  }

  // D layout: n = lane%16, m = r + 8*(lane/16)
#pragma unroll
  for (int mt = 0; mt < 2; ++mt)
#pragma unroll
    for (int nt = 0; nt < 2; ++nt) {
      const int n = n0 + wn * 32 + nt * 16 + ln16;
      const float bn = bias ? bias[n] : 0.0f;
#pragma unroll
      for (int r = 0; r < 8; ++r) {
        const int m = m0 + wm * 32 + mt * 16 + half * 8 + r;
        float vv = acc[mt][nt][r] + bn;
        if (act == 1) vv = gelu_exact(vv);
        if (Cf) Cf[(size_t)m * N + n] = vv;
        else    Cb[(size_t)m * N + n] = f2bf(vv);
      }
    }
}

// fp32 -> bf16 bulk convert (4 elems/thread)
__global__ __launch_bounds__(256) void cvt_bf16_k(
    const float* __restrict__ src, unsigned short* __restrict__ dst, int n4)
{
  int i = blockIdx.x * 256 + threadIdx.x;
  if (i >= n4) return;
  float4 v = ((const float4*)src)[i];
  unsigned long long p = (unsigned long long)f2bf(v.x)
                       | ((unsigned long long)f2bf(v.y) << 16)
                       | ((unsigned long long)f2bf(v.z) << 32)
                       | ((unsigned long long)f2bf(v.w) << 48);
  ((unsigned long long*)dst)[i] = p;
}

// Gather question tokens + LayerNorm -> bf16 Xln [B*Q, H]
__global__ __launch_bounds__(256) void gather_ln_k(
    const float* __restrict__ hs, const int* __restrict__ qidx,
    const float* __restrict__ g, const float* __restrict__ be,
    unsigned short* __restrict__ out)
{
  __shared__ float red[256];
  const int row = blockIdx.x;
  const int bi = row / Q_;
  const int qi = row - bi * Q_;
  const int src = qidx[bi * Q_ + qi];
  const float* x = hs + ((size_t)bi * S_ + src) * H_;
  const int t = threadIdx.x;

  float s = 0.f;
  for (int i = t; i < H_; i += 256) s += x[i];
  float mu = block_reduce_sum(s, red) * (1.0f / H_);

  float v = 0.f;
  for (int i = t; i < H_; i += 256) { float d = x[i] - mu; v += d * d; }
  float var = block_reduce_sum(v, red) * (1.0f / H_);
  float rs = rsqrtf(var + 1e-5f);

  unsigned short* o = out + (size_t)row * H_;
  for (int i = t; i < H_; i += 256) o[i] = f2bf((x[i] - mu) * rs * g[i] + be[i]);
}

// Attention 1: Qp fp32 [SEL,H] (batch-invariant), K1/V1 bf16 [B*Q,H], 512 keys
__global__ __launch_bounds__(256) void attn_qa_k(
    const float* __restrict__ Qp, const unsigned short* __restrict__ K1,
    const unsigned short* __restrict__ V1, unsigned short* __restrict__ O)
{
  __shared__ float qv[D_];
  __shared__ float sc[Q_];
  __shared__ float red[256];
  const int q = blockIdx.x, h = blockIdx.y, b = blockIdx.z;
  const int t = threadIdx.x;

  qv[t] = Qp[(size_t)q * H_ + h * D_ + t];
  __syncthreads();

  const float scale = 0.0625f;
  float lmax = -3.4e38f;
  for (int kk = t; kk < Q_; kk += 256) {
    const unsigned short* kp = K1 + ((size_t)(b * Q_ + kk)) * H_ + h * D_;
    float s = 0.f;
    for (int d = 0; d < D_; ++d) s += qv[d] * bf2f(kp[d]);
    s *= scale;
    sc[kk] = s;
    lmax = fmaxf(lmax, s);
  }
  __syncthreads();
  float mx = block_reduce_max(lmax, red);
  float lsum = 0.f;
  for (int kk = t; kk < Q_; kk += 256) { float p = __expf(sc[kk] - mx); sc[kk] = p; lsum += p; }
  __syncthreads();
  float inv = 1.0f / block_reduce_sum(lsum, red);
  __syncthreads();

  float o = 0.f;
  for (int kk = 0; kk < Q_; ++kk)
    o += sc[kk] * bf2f(V1[((size_t)(b * Q_ + kk)) * H_ + h * D_ + t]);
  O[((size_t)(b * SEL_ + q)) * H_ + h * D_ + t] = f2bf(o * inv);
}

// Attention 2: Q2 fp32 rows b*SEL.., K2/V2 bf16 [CB,H] (batch-invariant), 2048 keys
__global__ __launch_bounds__(256) void attn_cb_k(
    const float* __restrict__ Q2, const unsigned short* __restrict__ K2,
    const unsigned short* __restrict__ V2, unsigned short* __restrict__ O)
{
  __shared__ float qv[D_];
  __shared__ float sc[CB_];
  __shared__ float red[256];
  const int q = blockIdx.x, h = blockIdx.y, b = blockIdx.z;
  const int t = threadIdx.x;

  qv[t] = Q2[((size_t)(b * SEL_ + q)) * H_ + h * D_ + t];
  __syncthreads();

  const float scale = 0.0625f;
  float lmax = -3.4e38f;
  for (int kk = t; kk < CB_; kk += 256) {
    const unsigned short* kp = K2 + (size_t)kk * H_ + h * D_;
    float s = 0.f;
    for (int d = 0; d < D_; ++d) s += qv[d] * bf2f(kp[d]);
    s *= scale;
    sc[kk] = s;
    lmax = fmaxf(lmax, s);
  }
  __syncthreads();
  float mx = block_reduce_max(lmax, red);
  float lsum = 0.f;
  for (int kk = t; kk < CB_; kk += 256) { float p = __expf(sc[kk] - mx); sc[kk] = p; lsum += p; }
  __syncthreads();
  float inv = 1.0f / block_reduce_sum(lsum, red);
  __syncthreads();

  float o = 0.f;
  for (int kk = 0; kk < CB_; ++kk)
    o += sc[kk] * bf2f(V2[(size_t)kk * H_ + h * D_ + t]);
  O[((size_t)(b * SEL_ + q)) * H_ + h * D_ + t] = f2bf(o * inv);
}

// final_q = base + 0.3*(qaOut - base), written bf16
__global__ __launch_bounds__(256) void finalq_k(
    const float* __restrict__ qaOut, const float* __restrict__ lq,
    unsigned short* __restrict__ fq, int total)
{
  int i = blockIdx.x * 256 + threadIdx.x;
  if (i >= total) return;
  int r = i / H_;
  int s = r % SEL_;
  int c = i - r * H_;
  float base = lq[(size_t)s * H_ + c];
  fq[i] = f2bf(base + 0.3f * (qaOut[i] - base));
}

// RMSNorm + scatter into d_out at thinking_idx
__global__ __launch_bounds__(256) void rms_scatter_k(
    const float* __restrict__ tv, const float* __restrict__ w,
    const int* __restrict__ tidx, float* __restrict__ out)
{
  __shared__ float red[256];
  const int row = blockIdx.x;
  const int b = row / SEL_;
  const int s = row - b * SEL_;
  const float* x = tv + (size_t)row * H_;
  const int t = threadIdx.x;

  float ss = 0.f;
  for (int i = t; i < H_; i += 256) ss += x[i] * x[i];
  float rs = rsqrtf(block_reduce_sum(ss, red) * (1.0f / H_) + 1e-6f);

  float* dst = out + ((size_t)b * S_ + tidx[b * SEL_ + s]) * H_;
  for (int i = t; i < H_; i += 256) dst[i] = x[i] * rs * w[i];
}

// ---------------------------------------------------------------------------
extern "C" void kernel_launch(void* const* d_in, const int* in_sizes, int n_in,
                              void* d_out, int out_size, void* d_ws, size_t ws_size,
                              hipStream_t stream) {
  const float* hs    = (const float*)d_in[0];
  const float* ln_g  = (const float*)d_in[1];
  const float* ln_b  = (const float*)d_in[2];
  const float* w1    = (const float*)d_in[3];
  const float* b1    = (const float*)d_in[4];
  const float* w2    = (const float*)d_in[5];
  const float* b2    = (const float*)d_in[6];
  const float* lq    = (const float*)d_in[7];
  const float* qa_wi = (const float*)d_in[8];
  const float* qa_bi = (const float*)d_in[9];
  const float* qa_wo = (const float*)d_in[10];
  const float* qa_bo = (const float*)d_in[11];
  const float* cbk   = (const float*)d_in[12];
  const float* ca_wi = (const float*)d_in[13];
  const float* ca_bi = (const float*)d_in[14];
  const float* ca_wo = (const float*)d_in[15];
  const float* ca_bo = (const float*)d_in[16];
  const float* rms_w = (const float*)d_in[17];
  const int*   qidx  = (const int*)d_in[18];
  const int*   tidx  = (const int*)d_in[19];
  float* out = (float*)d_out;

  typedef unsigned short us;
  const size_t HH  = (size_t)H_ * H_;             // 4M elems
  char* p = (char*)d_ws;
  auto takeB = [&](size_t elems) { us* r = (us*)p; p += elems * 2; return r; };
  auto takeF = [&](size_t elems) { float* r = (float*)p; p += elems * 4; return r; };

  us* w1b    = takeB(HH);
  us* w2b    = takeB(HH);
  us* qa_wib = takeB(3 * HH);
  us* qa_wob = takeB(HH);
  us* ca_wib = takeB(3 * HH);
  us* ca_wob = takeB(HH);
  us* lqb    = takeB((size_t)SEL_ * H_);
  us* cbkb   = takeB((size_t)CB_ * H_);
  us* Xln    = takeB((size_t)B_ * Q_ * H_);       // reused as K2
  us* T1     = takeB((size_t)B_ * Q_ * H_);       // reused as V2
  us* clean  = takeB((size_t)B_ * Q_ * H_);
  us* K1     = takeB((size_t)B_ * Q_ * H_);
  us* V1     = takeB((size_t)B_ * Q_ * H_);
  float* Qp    = takeF((size_t)SEL_ * H_);
  us*    O1    = takeB((size_t)B_ * SEL_ * H_);
  float* qaOut = takeF((size_t)B_ * SEL_ * H_);
  us*    fq    = takeB((size_t)B_ * SEL_ * H_);
  float* Q2    = takeF((size_t)B_ * SEL_ * H_);
  us*    O2    = takeB((size_t)B_ * SEL_ * H_);
  float* tv    = takeF((size_t)B_ * SEL_ * H_);
  us* K2 = Xln;
  us* V2 = T1;

  // 0) passthrough copy
  hipMemcpyAsync(out, hs, (size_t)B_ * S_ * H_ * sizeof(float),
                 hipMemcpyDeviceToDevice, stream);

  // 1) one-time weight conversions to bf16
  auto cvt = [&](const float* s, us* d, size_t n) {
    int n4 = (int)(n / 4);
    cvt_bf16_k<<<(n4 + 255) / 256, 256, 0, stream>>>(s, d, n4);
  };
  cvt(w1, w1b, HH);           cvt(w2, w2b, HH);
  cvt(qa_wi, qa_wib, 3 * HH); cvt(qa_wo, qa_wob, HH);
  cvt(ca_wi, ca_wib, 3 * HH); cvt(ca_wo, ca_wob, HH);
  cvt(lq, lqb, (size_t)SEL_ * H_);
  cvt(cbk, cbkb, (size_t)CB_ * H_);

  // 2) gather + LayerNorm (bf16 out)
  gather_ln_k<<<B_ * Q_, 256, 0, stream>>>(hs, qidx, ln_g, ln_b, Xln);

  const int MQ = B_ * Q_;                 // 2048
  const int MS = B_ * SEL_;               // 256
  dim3 gBig(H_ / BNt, MQ / BM);           // 16 x 32
  dim3 gCB (H_ / BNt, CB_ / BM);          // 16 x 32
  dim3 gSm (H_ / BNt, MS / BM);           // 16 x 4
  dim3 gQp (H_ / BNt, SEL_ / BM);         // 16 x 1

  // 3) FFN
  wmma_gemm_bf16<<<gBig, 256, 0, stream>>>(Xln, w1b, b1, nullptr, T1, MQ, H_, H_, 1);
  wmma_gemm_bf16<<<gBig, 256, 0, stream>>>(T1, w2b, b2, nullptr, clean, MQ, H_, H_, 0);

  // 4) QA projections (Q-proj batch-invariant, once)
  wmma_gemm_bf16<<<gBig, 256, 0, stream>>>(clean, qa_wib + HH,     qa_bi + H_,     nullptr, K1, MQ, H_, H_, 0);
  wmma_gemm_bf16<<<gBig, 256, 0, stream>>>(clean, qa_wib + 2 * HH, qa_bi + 2 * H_, nullptr, V1, MQ, H_, H_, 0);
  wmma_gemm_bf16<<<gQp,  256, 0, stream>>>(lqb, qa_wib, qa_bi, Qp, nullptr, SEL_, H_, H_, 0);

  // 5) QA attention + out-proj + residual blend
  attn_qa_k<<<dim3(SEL_, NH_, B_), 256, 0, stream>>>(Qp, K1, V1, O1);
  wmma_gemm_bf16<<<gSm, 256, 0, stream>>>(O1, qa_wob, qa_bo, qaOut, nullptr, MS, H_, H_, 0);
  finalq_k<<<(MS * H_ + 255) / 256, 256, 0, stream>>>(qaOut, lq, fq, MS * H_);

  // 6) codebook attention (K/V batch-invariant, once)
  wmma_gemm_bf16<<<gSm, 256, 0, stream>>>(fq, ca_wib, ca_bi, Q2, nullptr, MS, H_, H_, 0);
  wmma_gemm_bf16<<<gCB, 256, 0, stream>>>(cbkb, ca_wib + HH,     ca_bi + H_,     nullptr, K2, CB_, H_, H_, 0);
  wmma_gemm_bf16<<<gCB, 256, 0, stream>>>(cbkb, ca_wib + 2 * HH, ca_bi + 2 * H_, nullptr, V2, CB_, H_, H_, 0);
  attn_cb_k<<<dim3(SEL_, NH_, B_), 256, 0, stream>>>(Q2, K2, V2, O2);
  wmma_gemm_bf16<<<gSm, 256, 0, stream>>>(O2, ca_wob, ca_bo, tv, nullptr, MS, H_, H_, 0);

  // 7) RMSNorm + scatter
  rms_scatter_k<<<MS, 256, 0, stream>>>(tv, rms_w, tidx, out);
}
